// GraphGDP_42580305773046
// MI455X (gfx1250) — matrix operand order repaired and checked by reading, not compiled, not run on hardware
//
#include <hip/hip_runtime.h>
#include <hip/hip_bf16.h>
#include <math.h>

// ---------------------------------------------------------------------------
// Types for CDNA5 WMMA (wave32, v_wmma_f32_16x16x32_bf16)
// ---------------------------------------------------------------------------
typedef __bf16 bf16;
typedef bf16  v16bf __attribute__((ext_vector_type(16)));
typedef bf16  v8bf  __attribute__((ext_vector_type(8)));
typedef float v8f   __attribute__((ext_vector_type(8)));

#define NNODES 10000
#define E1N    320000
#define E2N    160000
#define HD     256
#define HD2    512
#define NLAYER 3
#define NEG_SLOPE 0.2f

// ---------------------------------------------------------------------------
// Weight pre-pass: W (K,Ncols) fp32 row-major  ->  Wt (Ncols,K) bf16 row-major
// ---------------------------------------------------------------------------
__global__ void convert_transpose_w(const float* __restrict__ W,
                                    bf16* __restrict__ Wt, int K, int Ncols)
{
    long i = (long)blockIdx.x * blockDim.x + threadIdx.x;
    if (i >= (long)K * Ncols) return;
    int n = (int)(i % Ncols);
    int k = (int)(i / Ncols);
    Wt[(long)n * K + k] = (bf16)W[i];
}

// ---------------------------------------------------------------------------
// bf16 WMMA GEMM:  C = act(A(M,K) @ W(K,Ncols) + bias)
//   A : bf16 row-major (stride K); decoder variant gathers rows:
//       cols [0,512) from A[gS[r]*512+k], cols [512,1024) from A[gD[r]*512+..]
//       (handled as two branchless k-phases)
//   Bt: transposed bf16 weights (Ncols,K) row-major
// One wave computes a 32x64 tile: 2 m-subtiles x 4 n-subtiles = 8 accumulators
// sharing 4 B fragments and 2 A fragments per k-step (8 WMMA / load batch).
// 8 waves/block. M%16==0, K%32==0, Ncols%64==0.
// Lane mappings (hi = lane>>4; A/B use contiguous k order, this code's choice, not the ISA table):
//   A frag: lane m=l&15; halves = A[m][k0+hi*16 .. +16)
//   B frag: lane n=l&15; halves = B[k0+hi*16 .. +16)[n]  (contiguous in Bt)
//   C frag: lane n=l&15; float j -> row m = hi*8 + j
// ---------------------------------------------------------------------------
struct Acc8 { v8f a[8]; };

__device__ __forceinline__ void wmma_kloop(const bf16* __restrict__ r0,
                                           const bf16* __restrict__ r1,
                                           const bf16* __restrict__ b0p,
                                           const bf16* __restrict__ b1p,
                                           const bf16* __restrict__ b2p,
                                           const bf16* __restrict__ b3p,
                                           int kLen, int hi, Acc8& acc)
{
    for (int k0 = 0; k0 < kLen; k0 += 32) {
        union { v16bf v; v8bf h[2]; } A0, A1, B0, B1, B2, B3;
        const int ka0 = k0 + hi * 16;
        const int ka1 = k0 + hi * 16 + 8;
        A0.h[0] = *(const v8bf*)(r0 + ka0);
        A0.h[1] = *(const v8bf*)(r0 + ka1);
        A1.h[0] = *(const v8bf*)(r1 + ka0);
        A1.h[1] = *(const v8bf*)(r1 + ka1);
        const int kb = k0 + hi * 16;
        B0.v = *(const v16bf*)(b0p + kb);
        B1.v = *(const v16bf*)(b1p + kb);
        B2.v = *(const v16bf*)(b2p + kb);
        B3.v = *(const v16bf*)(b3p + kb);
        acc.a[0] = __builtin_amdgcn_wmma_f32_16x16x32_bf16(false, A0.v, false, B0.v, (short)0, acc.a[0], false, false);
        acc.a[1] = __builtin_amdgcn_wmma_f32_16x16x32_bf16(false, A0.v, false, B1.v, (short)0, acc.a[1], false, false);
        acc.a[2] = __builtin_amdgcn_wmma_f32_16x16x32_bf16(false, A0.v, false, B2.v, (short)0, acc.a[2], false, false);
        acc.a[3] = __builtin_amdgcn_wmma_f32_16x16x32_bf16(false, A0.v, false, B3.v, (short)0, acc.a[3], false, false);
        acc.a[4] = __builtin_amdgcn_wmma_f32_16x16x32_bf16(false, A1.v, false, B0.v, (short)0, acc.a[4], false, false);
        acc.a[5] = __builtin_amdgcn_wmma_f32_16x16x32_bf16(false, A1.v, false, B1.v, (short)0, acc.a[5], false, false);
        acc.a[6] = __builtin_amdgcn_wmma_f32_16x16x32_bf16(false, A1.v, false, B2.v, (short)0, acc.a[6], false, false);
        acc.a[7] = __builtin_amdgcn_wmma_f32_16x16x32_bf16(false, A1.v, false, B3.v, (short)0, acc.a[7], false, false);
    }
}

__device__ __forceinline__ void store_tile(v8f acc, int nc, long mbase, int hi,
                                           const float* __restrict__ bias,
                                           float* __restrict__ Cf,
                                           bf16* __restrict__ Cb,
                                           bf16* __restrict__ Cb2,
                                           int ldc, int relu)
{
    float bv = bias ? bias[nc] : 0.0f;
#pragma unroll
    for (int j = 0; j < 8; ++j) {
        long r = mbase + hi * 8 + j;
        float v = acc[j] + bv;
        if (relu) v = fmaxf(v, 0.0f);
        long o = r * (long)ldc + nc;
        if (Cf)  Cf[o]  = v;
        if (Cb)  Cb[o]  = (bf16)v;
        if (Cb2) Cb2[o] = (bf16)v;
    }
}

__global__ __launch_bounds__(256) void wmma_gemm_bf16(
    const bf16* __restrict__ A,
    const int* __restrict__ gS, const int* __restrict__ gD,
    const bf16* __restrict__ Bt, const float* __restrict__ bias,
    float* __restrict__ Cf, bf16* __restrict__ Cb, bf16* __restrict__ Cb2,
    int ldc, int M, int K, int Ncols, int relu)
{
    const int lane = threadIdx.x & 31;
    const int wave = threadIdx.x >> 5;
    const int ntg  = Ncols >> 6;                 // 64-column groups
    const long tile = (long)blockIdx.x * 8 + wave;
    const long tm   = tile / ntg;                // 32-row tile index
    const int  tg   = (int)(tile % ntg);
    if (tm * 32 >= (long)M) return;
    const bool hasM2 = (tm * 32 + 16) < (long)M; // second m-subtile in range?

    const int hi    = lane >> 4;
    const int mrow0 = (int)tm * 32 + (lane & 15);
    const int mrow1 = hasM2 ? (mrow0 + 16) : mrow0;
    const int ncol0 = tg * 64 + (lane & 15);

    const bf16* bt0 = Bt + (long)(ncol0)      * K;
    const bf16* bt1 = Bt + (long)(ncol0 + 16) * K;
    const bf16* bt2 = Bt + (long)(ncol0 + 32) * K;
    const bf16* bt3 = Bt + (long)(ncol0 + 48) * K;

    Acc8 acc;
#pragma unroll
    for (int i = 0; i < 8; ++i)
        acc.a[i] = (v8f){0.f,0.f,0.f,0.f,0.f,0.f,0.f,0.f};

    if (gS) {
        // decoder gather: K = 1024, row = [x1[gS[r]], x1[gD[r]]]
        const bf16* rS0 = A + (long)gS[mrow0] * 512;
        const bf16* rS1 = A + (long)gS[mrow1] * 512;
        const bf16* rD0 = A + (long)gD[mrow0] * 512;
        const bf16* rD1 = A + (long)gD[mrow1] * 512;
        __builtin_prefetch(rS0, 0, 1);
        __builtin_prefetch(rD0, 0, 1);
        wmma_kloop(rS0, rS1, bt0,       bt1,       bt2,       bt3,       512, hi, acc);
        wmma_kloop(rD0, rD1, bt0 + 512, bt1 + 512, bt2 + 512, bt3 + 512, 512, hi, acc);
    } else {
        const bf16* r0 = A + (long)mrow0 * K;
        const bf16* r1 = A + (long)mrow1 * K;
        __builtin_prefetch(r0, 0, 1);
        wmma_kloop(r0, r1, bt0, bt1, bt2, bt3, K, hi, acc);
    }

    const long mb0 = tm * 32;
    store_tile(acc.a[0], ncol0,      mb0, hi, bias, Cf, Cb, Cb2, ldc, relu);
    store_tile(acc.a[1], ncol0 + 16, mb0, hi, bias, Cf, Cb, Cb2, ldc, relu);
    store_tile(acc.a[2], ncol0 + 32, mb0, hi, bias, Cf, Cb, Cb2, ldc, relu);
    store_tile(acc.a[3], ncol0 + 48, mb0, hi, bias, Cf, Cb, Cb2, ldc, relu);
    if (hasM2) {
        const long mb1 = mb0 + 16;
        store_tile(acc.a[4], ncol0,      mb1, hi, bias, Cf, Cb, Cb2, ldc, relu);
        store_tile(acc.a[5], ncol0 + 16, mb1, hi, bias, Cf, Cb, Cb2, ldc, relu);
        store_tile(acc.a[6], ncol0 + 32, mb1, hi, bias, Cf, Cb, Cb2, ldc, relu);
        store_tile(acc.a[7], ncol0 + 48, mb1, hi, bias, Cf, Cb, Cb2, ldc, relu);
    }
}

// ---------------------------------------------------------------------------
// Small helper kernels
// ---------------------------------------------------------------------------
__global__ void fill_f32(float* __restrict__ p, float v, long n) {
    long i = (long)blockIdx.x * blockDim.x + threadIdx.x;
    if (i < n) p[i] = v;
}

// h0 = relu(t_value[batch[n]] * W0 + b0)  -> bf16 (N,256).  grid=N, block=256
__global__ void time_encode_l0(const int* __restrict__ batch,
                               const float* __restrict__ tval,
                               const float* __restrict__ W0,
                               const float* __restrict__ b0,
                               bf16* __restrict__ h0)
{
    int n = blockIdx.x, c = threadIdx.x;
    float t = tval[batch[n]];
    float v = fmaxf(t * W0[c] + b0[c], 0.0f);
    h0[(long)n * HD + c] = (bf16)v;
}

// h0 = relu(attr[e] * W0 + b0) -> bf16 (E,256).  grid=E, block=256
__global__ void edge_encode_l0(const float* __restrict__ attr,
                               const float* __restrict__ W0,
                               const float* __restrict__ b0,
                               bf16* __restrict__ h0)
{
    int e = blockIdx.x, c = threadIdx.x;
    float a = attr[e];
    float v = fmaxf(a * W0[c] + b0[c], 0.0f);
    h0[(long)e * HD + c] = (bf16)v;
}

__device__ __forceinline__ void atomicMaxF32(float* addr, float value) {
    int* ia = (int*)addr;
    int old = *ia;
    while (true) {
        float fo = __int_as_float(old);
        if (fo >= value) break;
        int assumed = old;
        old = atomicCAS(ia, assumed, __float_as_int(value));
        if (old == assumed) break;
    }
}

// warp-per-edge: logit = dot(leaky_relu(xl[s]+xr[d]+mW[e]), att); atomicMax lmax[d]
__global__ __launch_bounds__(256) void edge_logits_kernel(
    const float* __restrict__ xl, const float* __restrict__ xr,
    const float* __restrict__ mW,
    const int* __restrict__ src, const int* __restrict__ dst,
    const float* __restrict__ att,
    float* __restrict__ logit, float* __restrict__ lmax, int E)
{
    int e    = blockIdx.x * 8 + (threadIdx.x >> 5);
    int lane = threadIdx.x & 31;
    if (e >= E) return;
    int s = src[e], d = dst[e];
    const float* xls = xl + (long)s * HD;
    const float* xrd = xr + (long)d * HD;
    const float* mwe = mW + (long)e * HD;
    float sum = 0.f;
#pragma unroll
    for (int i = 0; i < 8; ++i) {
        int c = lane + i * 32;
        float m = xls[c] + xrd[c] + mwe[c];
        m = (m > 0.f) ? m : NEG_SLOPE * m;
        sum += m * att[c];
    }
    for (int off = 16; off; off >>= 1) sum += __shfl_down(sum, off, 32);
    if (lane == 0) {
        logit[e] = sum;
        atomicMaxF32(&lmax[d], sum);
    }
}

// ex = exp(logit - lmax[dst]); den[dst] += ex   (logit overwritten in place)
__global__ void edge_exp_kernel(float* __restrict__ logit,
                                const float* __restrict__ lmax,
                                const int* __restrict__ dst,
                                float* __restrict__ den, int E)
{
    int e = blockIdx.x * blockDim.x + threadIdx.x;
    if (e >= E) return;
    int d = dst[e];
    float lm = lmax[d];
    if (isinf(lm)) lm = 0.0f;
    float ex = expf(logit[e] - lm);
    logit[e] = ex;
    atomicAdd(&den[d], ex);
}

// agg[dst] += (ex/den[dst]) * xl[src]   grid=E, block=256 (one channel/thread)
__global__ void edge_aggregate_kernel(const float* __restrict__ ex,
                                      const float* __restrict__ den,
                                      const float* __restrict__ xl,
                                      const int* __restrict__ src,
                                      const int* __restrict__ dst,
                                      float* __restrict__ agg)
{
    int e = blockIdx.x, c = threadIdx.x;
    int s = src[e], d = dst[e];
    float alpha = ex[e] / (den[d] + 1e-16f);
    atomicAdd(&agg[(long)d * HD + c], alpha * xl[(long)s * HD + c]);
}

// out = relu(agg + b); write bf16 into two concat buffers (stride 512)
__global__ void finalize_node_kernel(const float* __restrict__ agg,
                                     const float* __restrict__ bias,
                                     bf16* __restrict__ o1, int off1,
                                     bf16* __restrict__ o2, int off2)
{
    int n = blockIdx.x, c = threadIdx.x;
    float v = fmaxf(agg[(long)n * HD + c] + bias[c], 0.0f);
    bf16 bv = (bf16)v;
    o1[(long)n * HD2 + off1 + c] = bv;
    o2[(long)n * HD2 + off2 + c] = bv;
}

// out[e] = dot(h[e,:], W2) + b2  (warp per row)
__global__ __launch_bounds__(256) void decoder_final_kernel(
    const bf16* __restrict__ h, const float* __restrict__ W2,
    const float* __restrict__ b2, float* __restrict__ out, int E)
{
    int e    = blockIdx.x * 8 + (threadIdx.x >> 5);
    int lane = threadIdx.x & 31;
    if (e >= E) return;
    const bf16* hr = h + (long)e * HD;
    float s = 0.f;
#pragma unroll
    for (int i = 0; i < 8; ++i) {
        int c = lane + i * 32;
        s += (float)hr[c] * W2[c];
    }
    for (int off = 16; off; off >>= 1) s += __shfl_down(s, off, 32);
    if (lane == 0) out[e] = s + b2[0];
}

// ---------------------------------------------------------------------------
// Host orchestration
// ---------------------------------------------------------------------------
static void launch_gemm(hipStream_t stream,
                        const bf16* A, const int* gS, const int* gD,
                        const bf16* Bt, const float* bias,
                        float* Cf, bf16* Cb, bf16* Cb2, int ldc,
                        int M, int K, int Ncols, int relu)
{
    int tiles  = ((M + 31) / 32) * (Ncols / 64);
    int blocks = (tiles + 7) / 8;
    wmma_gemm_bf16<<<blocks, 256, 0, stream>>>(A, gS, gD, Bt, bias, Cf, Cb, Cb2,
                                               ldc, M, K, Ncols, relu);
}

static void launch_fill(hipStream_t stream, float* p, float v, long n) {
    long blocks = (n + 255) / 256;
    fill_f32<<<(unsigned)blocks, 256, 0, stream>>>(p, v, n);
}

static void launch_conv(hipStream_t stream, const float* W, bf16* Wt, int K, int Ncols) {
    long n = (long)K * Ncols;
    long blocks = (n + 255) / 256;
    convert_transpose_w<<<(unsigned)blocks, 256, 0, stream>>>(W, Wt, K, Ncols);
}

extern "C" void kernel_launch(void* const* d_in, const int* in_sizes, int n_in,
                              void* d_out, int out_size, void* d_ws, size_t ws_size,
                              hipStream_t stream)
{
    (void)in_sizes; (void)n_in; (void)out_size; (void)ws_size;

    const int*   ei1   = (const int*)d_in[0];
    const float* ea1   = (const float*)d_in[1];
    const int*   ei2   = (const int*)d_in[2];
    const float* ea2   = (const float*)d_in[3];
    const int*   batch = (const int*)d_in[4];
    const float* tval  = (const float*)d_in[5];
    const float* te_W0 = (const float*)d_in[6],  *te_b0 = (const float*)d_in[7];
    const float* te_W1 = (const float*)d_in[8],  *te_b1 = (const float*)d_in[9];
    const float* te_W2 = (const float*)d_in[10], *te_b2 = (const float*)d_in[11];
    const float* ee_W0 = (const float*)d_in[12], *ee_b0 = (const float*)d_in[13];
    const float* ee_W1 = (const float*)d_in[14], *ee_b1 = (const float*)d_in[15];
    const float* ee_W2 = (const float*)d_in[16], *ee_b2 = (const float*)d_in[17];
    const float* de_W0 = (const float*)d_in[18], *de_b0 = (const float*)d_in[19];
    const float* de_W1 = (const float*)d_in[20], *de_b1 = (const float*)d_in[21];
    const float* de_W2 = (const float*)d_in[22], *de_b2 = (const float*)d_in[23];
    const float* gg_Wl = (const float*)d_in[24], *gg_Wr = (const float*)d_in[25];
    const float* gg_We = (const float*)d_in[26], *gg_att= (const float*)d_in[27];
    const float* gg_b  = (const float*)d_in[28];
    const float* gf_Wl = (const float*)d_in[29], *gf_Wr = (const float*)d_in[30];
    const float* gf_We = (const float*)d_in[31], *gf_att= (const float*)d_in[32];
    const float* gf_b  = (const float*)d_in[33];

    const int* s1 = ei1;             const int* t1 = ei1 + E1N;
    const int* s2 = ei2;             const int* t2 = ei2 + E2N;

    // ---- workspace carve-up -------------------------------------------------
    char* w = (char*)d_ws;
    size_t off = 0;
    auto take = [&](size_t bytes) -> char* {
        char* p = w + off;
        off = (off + bytes + 255) & ~(size_t)255;
        return p;
    };
    // 328MB region, time-shared: (a) two bf16 ping/pong bufs for MLPs,
    // (b) fp32 mW = ee @ We during GAT layers.
    float* mW   = (float*)take((size_t)E1N * HD * sizeof(float));
    bf16*  bufA = (bf16*)mW;
    bf16*  bufB = (bf16*)((char*)mW + (size_t)E1N * HD * sizeof(bf16));

    bf16* ee1b = (bf16*)take((size_t)E1N * HD * sizeof(bf16));
    bf16* ee2b = (bf16*)take((size_t)E2N * HD * sizeof(bf16));
    bf16* P[4];
    for (int i = 0; i < 4; ++i) P[i] = (bf16*)take((size_t)NNODES * HD2 * sizeof(bf16));
    float* xl   = (float*)take((size_t)NNODES * HD * sizeof(float));
    float* xr   = (float*)take((size_t)NNODES * HD * sizeof(float));
    float* agg  = (float*)take((size_t)NNODES * HD * sizeof(float));
    float* elog = (float*)take((size_t)E1N * sizeof(float));
    float* lmax = (float*)take((size_t)NNODES * sizeof(float));
    float* den  = (float*)take((size_t)NNODES * sizeof(float));

    // transposed bf16 weights (~5.2 MB total)
    const size_t SQ = (size_t)HD * HD;          // 256x256
    bf16* wt_te1 = (bf16*)take(SQ * 2);
    bf16* wt_te2 = (bf16*)take(SQ * 2);
    bf16* wt_ee1 = (bf16*)take(SQ * 2);
    bf16* wt_ee2 = (bf16*)take(SQ * 2);
    bf16* wt_de0 = (bf16*)take((size_t)HD * 4 * HD * 2);   // 256 x 1024
    bf16* wt_de1 = (bf16*)take(SQ * 2);
    bf16* wt_Wl[2][NLAYER];
    bf16* wt_Wr[2][NLAYER];
    bf16* wt_We[2][NLAYER];
    for (int g = 0; g < 2; ++g)
        for (int l = 0; l < NLAYER; ++l) {
            wt_Wl[g][l] = (bf16*)take((size_t)HD * HD2 * 2); // 256 x 512
            wt_Wr[g][l] = (bf16*)take((size_t)HD * HD2 * 2);
            wt_We[g][l] = (bf16*)take(SQ * 2);
        }

    // ---- weight conversion pre-pass ----------------------------------------
    launch_conv(stream, te_W1, wt_te1, HD, HD);
    launch_conv(stream, te_W2, wt_te2, HD, HD);
    launch_conv(stream, ee_W1, wt_ee1, HD, HD);
    launch_conv(stream, ee_W2, wt_ee2, HD, HD);
    launch_conv(stream, de_W0, wt_de0, 4 * HD, HD);
    launch_conv(stream, de_W1, wt_de1, HD, HD);
    for (int l = 0; l < NLAYER; ++l) {
        launch_conv(stream, gg_Wl + (size_t)l * HD2 * HD, wt_Wl[0][l], HD2, HD);
        launch_conv(stream, gg_Wr + (size_t)l * HD2 * HD, wt_Wr[0][l], HD2, HD);
        launch_conv(stream, gg_We + (size_t)l * HD * HD,  wt_We[0][l], HD,  HD);
        launch_conv(stream, gf_Wl + (size_t)l * HD2 * HD, wt_Wl[1][l], HD2, HD);
        launch_conv(stream, gf_Wr + (size_t)l * HD2 * HD, wt_Wr[1][l], HD2, HD);
        launch_conv(stream, gf_We + (size_t)l * HD * HD,  wt_We[1][l], HD,  HD);
    }

    // ---- encoders (bufA/bufB phase) ----------------------------------------
    // time encoder -> x_init = concat([t_enc, t_enc]) in P[0]
    time_encode_l0<<<NNODES, HD, 0, stream>>>(batch, tval, te_W0, te_b0, bufA);
    launch_gemm(stream, bufA, nullptr, nullptr, wt_te1, te_b1,
                nullptr, bufB, nullptr, HD, NNODES, HD, HD, 1);
    launch_gemm(stream, bufB, nullptr, nullptr, wt_te2, te_b2,
                nullptr, P[0], P[0] + HD, HD2, NNODES, HD, HD, 0);

    // edge encoders (shared ee weights)
    edge_encode_l0<<<E1N, HD, 0, stream>>>(ea1, ee_W0, ee_b0, bufA);
    launch_gemm(stream, bufA, nullptr, nullptr, wt_ee1, ee_b1,
                nullptr, bufB, nullptr, HD, E1N, HD, HD, 1);
    launch_gemm(stream, bufB, nullptr, nullptr, wt_ee2, ee_b2,
                nullptr, ee1b, nullptr, HD, E1N, HD, HD, 0);

    edge_encode_l0<<<E2N, HD, 0, stream>>>(ea2, ee_W0, ee_b0, bufA);
    launch_gemm(stream, bufA, nullptr, nullptr, wt_ee1, ee_b1,
                nullptr, bufB, nullptr, HD, E2N, HD, HD, 1);
    launch_gemm(stream, bufB, nullptr, nullptr, wt_ee2, ee_b2,
                nullptr, ee2b, nullptr, HD, E2N, HD, HD, 0);

    // ---- GAT layers (mW phase) ---------------------------------------------
    bf16 *in1 = P[0], *in2 = P[0];
    for (int l = 0; l < NLAYER; ++l) {
        bf16* out1 = (l == 1) ? P[3] : P[1];
        bf16* out2 = (l == 1) ? P[0] : P[2];
        const size_t oV = (size_t)l * HD;

        // ---- graph 1 (gg params) ----
        launch_gemm(stream, in1, nullptr, nullptr, wt_Wl[0][l], nullptr,
                    xl, nullptr, nullptr, HD, NNODES, HD2, HD, 0);
        launch_gemm(stream, in1, nullptr, nullptr, wt_Wr[0][l], nullptr,
                    xr, nullptr, nullptr, HD, NNODES, HD2, HD, 0);
        launch_gemm(stream, ee1b, nullptr, nullptr, wt_We[0][l], nullptr,
                    mW, nullptr, nullptr, HD, E1N, HD, HD, 0);
        launch_fill(stream, lmax, -INFINITY, NNODES);
        launch_fill(stream, den, 0.0f, NNODES);
        launch_fill(stream, agg, 0.0f, (long)NNODES * HD);
        edge_logits_kernel<<<(E1N + 7) / 8, 256, 0, stream>>>(
            xl, xr, mW, s1, t1, gg_att + oV, elog, lmax, E1N);
        edge_exp_kernel<<<(E1N + 255) / 256, 256, 0, stream>>>(elog, lmax, t1, den, E1N);
        edge_aggregate_kernel<<<E1N, HD, 0, stream>>>(elog, den, xl, s1, t1, agg);
        // o1 -> x1_next[:,256:512], x2_next[:,0:256]
        finalize_node_kernel<<<NNODES, HD, 0, stream>>>(agg, gg_b + oV, out1, HD, out2, 0);

        // ---- graph 2 (gf params) ----
        launch_gemm(stream, in2, nullptr, nullptr, wt_Wl[1][l], nullptr,
                    xl, nullptr, nullptr, HD, NNODES, HD2, HD, 0);
        launch_gemm(stream, in2, nullptr, nullptr, wt_Wr[1][l], nullptr,
                    xr, nullptr, nullptr, HD, NNODES, HD2, HD, 0);
        launch_gemm(stream, ee2b, nullptr, nullptr, wt_We[1][l], nullptr,
                    mW, nullptr, nullptr, HD, E2N, HD, HD, 0);
        launch_fill(stream, lmax, -INFINITY, NNODES);
        launch_fill(stream, den, 0.0f, NNODES);
        launch_fill(stream, agg, 0.0f, (long)NNODES * HD);
        edge_logits_kernel<<<(E2N + 7) / 8, 256, 0, stream>>>(
            xl, xr, mW, s2, t2, gf_att + oV, elog, lmax, E2N);
        edge_exp_kernel<<<(E2N + 255) / 256, 256, 0, stream>>>(elog, lmax, t2, den, E2N);
        edge_aggregate_kernel<<<E2N, HD, 0, stream>>>(elog, den, xl, s2, t2, agg);
        // o2 -> x1_next[:,0:256], x2_next[:,256:512]
        finalize_node_kernel<<<NNODES, HD, 0, stream>>>(agg, gf_b + oV, out1, 0, out2, HD);

        in1 = out1;
        in2 = out2;
    }

    // ---- decoder (bufA/bufB phase again) -----------------------------------
    // layer 0: gathered A rows = [x1[s1], x1[d1]] (K=1024), two branchless
    // k-phases inside the GEMM
    launch_gemm(stream, in1, s1, t1, wt_de0, de_b0,
                nullptr, bufA, nullptr, HD, E1N, 4 * HD, HD, 1);
    launch_gemm(stream, bufA, nullptr, nullptr, wt_de1, de_b1,
                nullptr, bufB, nullptr, HD, E1N, HD, HD, 1);
    decoder_final_kernel<<<(E1N + 7) / 8, 256, 0, stream>>>(
        bufB, de_W2, de_b2, (float*)d_out, E1N);
}
